// DualModalVQTokenizer_68161130988020
// MI455X (gfx1250) — compile-verified, matrix operand order checked by reading, not run on hardware
//
#include <hip/hip_runtime.h>
#include <math.h>

typedef unsigned short u16;
typedef __attribute__((ext_vector_type(16))) __bf16 v16bf;
typedef __attribute__((ext_vector_type(8)))  float  v8f;

union AFrag { unsigned int u[8]; v16bf v; };

// ---------------- small device helpers ----------------

__device__ __forceinline__ float geluf(float x) {
    return 0.5f * x * (1.0f + erff(x * 0.70710678118654752f));
}
__device__ __forceinline__ float sigm(float x) { return 1.0f / (1.0f + expf(-x)); }

__device__ __forceinline__ u16 f2bf(float f) {
    unsigned int u = __builtin_bit_cast(unsigned int, f);
    u = (u + 0x7FFFu + ((u >> 16) & 1u)) >> 16;
    return (u16)u;
}

// CDNA5 async memory->LDS copy (per-lane 16B), tracked by ASYNCcnt.
// LDS address taken from the VDST operand; the low 32 bits of a generic pointer to a
// __shared__ object are the wave-relative LDS byte offset.
__device__ __forceinline__ void async_load_b128(void* lds_ptr, const void* gptr) {
    unsigned lds_off = (unsigned)(size_t)lds_ptr;
    asm volatile("global_load_async_to_lds_b128 %0, %1, off"
                 :: "v"(lds_off), "v"(gptr) : "memory");
}
__device__ __forceinline__ void wait_async0() {
    asm volatile("s_wait_asynccnt 0" ::: "memory");
}

// block = 256 threads (8 wave32). Sum s and q across the block.
__device__ __forceinline__ void block_sum2(float& s, float& q) {
    #pragma unroll
    for (int off = 16; off > 0; off >>= 1) {
        s += __shfl_xor(s, off, 32);
        q += __shfl_xor(q, off, 32);
    }
    __shared__ float sm[8], qm[8];
    int w = threadIdx.x >> 5, l = threadIdx.x & 31;
    if (l == 0) { sm[w] = s; qm[w] = q; }
    __syncthreads();
    if (threadIdx.x == 0) {
        float a = 0.f, b = 0.f;
        #pragma unroll
        for (int i = 0; i < 8; i++) { a += sm[i]; b += qm[i]; }
        sm[0] = a; qm[0] = b;
    }
    __syncthreads();
    s = sm[0]; q = qm[0];
}

// Build A/B fragments per the CDNA5 16-bit WMMA VGPR layout and issue one
// v_wmma_f32_16x16x32_bf16.  A row = lane&15 (wave half selects K+8 block),
// B col = lane&15 symmetric.  LDS tiles are row-major with leading dim ld*.
__device__ __forceinline__ v8f wmma_bf16_frag(const u16* As, int lda,
                                              const u16* Ws, int ldb,
                                              int kc, int lane,
                                              int waveM, int waveN, v8f acc) {
    const int half = lane >> 4, l16 = lane & 15;
    AFrag a, b;
    #pragma unroll
    for (int j = 0; j < 8; j++) {
        int kb = kc + ((j >> 2) << 4) + (half << 3) + ((j & 3) << 1);
        a.u[j] = *(const unsigned int*)(As + (waveM * 16 + l16) * lda + kb);
        b.u[j] = *(const unsigned int*)(Ws + (waveN * 16 + l16) * ldb + kb);
    }
    return __builtin_amdgcn_wmma_f32_16x16x32_bf16(false, a.v, false, b.v,
                                                   (short)0, acc, false, false);
}

// ---------------- init ----------------

__global__ void init_loss_kernel(float* loss) { *loss = 0.0f; }

__global__ void f32_to_bf16_kernel(const float* __restrict__ src, u16* __restrict__ dst, int n) {
    int i = blockIdx.x * 256 + threadIdx.x;
    if (i < n) dst[i] = f2bf(src[i]);
}

// ---------------- motion encoder (conv1d + GN + GELU) ----------------
// conv1: [B,T,8] -> [B,32,4096], k7 s1 p3, GN groups=8 (4 ch/group)
__global__ void me_conv1_kernel(const float* __restrict__ x, const float* __restrict__ w,
                                const float* __restrict__ bias, const float* __restrict__ gam,
                                const float* __restrict__ bet, float* __restrict__ out) {
    int b = blockIdx.x, g = blockIdx.y, tid = threadIdx.x;
    float vals[64]; float s = 0.f, q = 0.f;
    for (int i = 0; i < 64; i++) {
        int idx = i * 256 + tid;
        int cl = idx >> 12, t = idx & 4095;
        int co = g * 4 + cl;
        float acc = bias[co];
        for (int j = 0; j < 7; j++) {
            int tt = t + j - 3;
            if (tt >= 0 && tt < 4096) {
                const float* xr = x + ((size_t)b * 4096 + tt) * 8;
                const float* wr = w + co * 56 + j;
                #pragma unroll
                for (int ci = 0; ci < 8; ci++) acc += xr[ci] * wr[ci * 7];
            }
        }
        vals[i] = acc; s += acc; q += acc * acc;
    }
    block_sum2(s, q);
    float mean = s * (1.f / 16384.f);
    float var  = q * (1.f / 16384.f) - mean * mean;
    float inv  = rsqrtf(var + 1e-5f);
    for (int i = 0; i < 64; i++) {
        int idx = i * 256 + tid;
        int cl = idx >> 12, t = idx & 4095;
        int co = g * 4 + cl;
        float v = (vals[i] - mean) * inv * gam[co] + bet[co];
        out[((size_t)b * 32 + co) * 4096 + t] = geluf(v);
    }
}

// conv2: [B,32,4096] -> [B,64,2048], k4 s2 p1, GN groups=8 (8 ch/group)
__global__ void me_conv2_kernel(const float* __restrict__ in, const float* __restrict__ w,
                                const float* __restrict__ bias, const float* __restrict__ gam,
                                const float* __restrict__ bet, float* __restrict__ out) {
    int b = blockIdx.x, g = blockIdx.y, tid = threadIdx.x;
    float vals[64]; float s = 0.f, q = 0.f;
    const float* ib = in + (size_t)b * 131072;  // 32*4096
    for (int i = 0; i < 64; i++) {
        int idx = i * 256 + tid;
        int cl = idx >> 11, t = idx & 2047;
        int co = g * 8 + cl;
        float acc = bias[co];
        for (int j = 0; j < 4; j++) {
            int tt = 2 * t + j - 1;
            if (tt >= 0 && tt < 4096) {
                const float* ir = ib + tt;
                const float* wr = w + co * 128 + j;
                #pragma unroll
                for (int ci = 0; ci < 32; ci++) acc += ir[ci * 4096] * wr[ci * 4];
            }
        }
        vals[i] = acc; s += acc; q += acc * acc;
    }
    block_sum2(s, q);
    float mean = s * (1.f / 16384.f);
    float var  = q * (1.f / 16384.f) - mean * mean;
    float inv  = rsqrtf(var + 1e-5f);
    for (int i = 0; i < 64; i++) {
        int idx = i * 256 + tid;
        int cl = idx >> 11, t = idx & 2047;
        int co = g * 8 + cl;
        float v = (vals[i] - mean) * inv * gam[co] + bet[co];
        out[((size_t)b * 64 + co) * 2048 + t] = geluf(v);
    }
}

// conv3: [B,64,2048] -> z_motion [B,1024,128], k4 s2 p1, GN groups=8 (16 ch/group)
__global__ void me_conv3_kernel(const float* __restrict__ in, const float* __restrict__ w,
                                const float* __restrict__ bias, const float* __restrict__ gam,
                                const float* __restrict__ bet, float* __restrict__ zmot) {
    int b = blockIdx.x, g = blockIdx.y, tid = threadIdx.x;
    float vals[64]; float s = 0.f, q = 0.f;
    const float* ib = in + (size_t)b * 131072;  // 64*2048
    for (int i = 0; i < 64; i++) {
        int idx = i * 256 + tid;
        int cl = idx >> 10, t = idx & 1023;
        int co = g * 16 + cl;
        float acc = bias[co];
        for (int j = 0; j < 4; j++) {
            int tt = 2 * t + j - 1;
            if (tt >= 0 && tt < 2048) {
                const float* ir = ib + tt;
                const float* wr = w + co * 256 + j;
                #pragma unroll
                for (int ci = 0; ci < 64; ci++) acc += ir[ci * 2048] * wr[ci * 4];
            }
        }
        vals[i] = acc; s += acc; q += acc * acc;
    }
    block_sum2(s, q);
    float mean = s * (1.f / 16384.f);
    float var  = q * (1.f / 16384.f) - mean * mean;
    float inv  = rsqrtf(var + 1e-5f);
    for (int i = 0; i < 64; i++) {
        int idx = i * 256 + tid;
        int cl = idx >> 10, t = idx & 1023;
        int co = g * 16 + cl;
        float v = (vals[i] - mean) * inv * gam[co] + bet[co];
        zmot[((size_t)b * 1024 + t) * 128 + co] = geluf(v);
    }
}

// ---------------- image encoder (conv2d + GN + GELU) ----------------
// ie1: [B,1,128,128] -> [B,16,64,64], GN groups=4 (4 ch/group)
__global__ void ie_conv1_kernel(const float* __restrict__ img, const float* __restrict__ w,
                                const float* __restrict__ bias, const float* __restrict__ gam,
                                const float* __restrict__ bet, float* __restrict__ out) {
    int b = blockIdx.x, g = blockIdx.y, tid = threadIdx.x;
    float vals[64]; float s = 0.f, q = 0.f;
    const float* ib = img + (size_t)b * 16384;
    for (int i = 0; i < 64; i++) {
        int idx = i * 256 + tid;
        int cl = idx >> 12, sp = idx & 4095;
        int oh = sp >> 6, ow = sp & 63;
        int co = g * 4 + cl;
        float acc = bias[co];
        #pragma unroll
        for (int kh = 0; kh < 4; kh++) {
            int ih = 2 * oh - 1 + kh;
            if (ih < 0 || ih >= 128) continue;
            #pragma unroll
            for (int kw = 0; kw < 4; kw++) {
                int iw = 2 * ow - 1 + kw;
                if (iw < 0 || iw >= 128) continue;
                acc += ib[ih * 128 + iw] * w[co * 16 + kh * 4 + kw];
            }
        }
        vals[i] = acc; s += acc; q += acc * acc;
    }
    block_sum2(s, q);
    float mean = s * (1.f / 16384.f);
    float var  = q * (1.f / 16384.f) - mean * mean;
    float inv  = rsqrtf(var + 1e-5f);
    for (int i = 0; i < 64; i++) {
        int idx = i * 256 + tid;
        int cl = idx >> 12, sp = idx & 4095;
        int co = g * 4 + cl;
        float v = (vals[i] - mean) * inv * gam[co] + bet[co];
        out[((size_t)b * 16 + co) * 4096 + sp] = geluf(v);
    }
}

// ie2: [B,16,64,64] -> [B,32,32,32], GN groups=8 (4 ch/group)
__global__ void ie_conv2_kernel(const float* __restrict__ in, const float* __restrict__ w,
                                const float* __restrict__ bias, const float* __restrict__ gam,
                                const float* __restrict__ bet, float* __restrict__ out) {
    int b = blockIdx.x, g = blockIdx.y, tid = threadIdx.x;
    float vals[16]; float s = 0.f, q = 0.f;
    const float* ib = in + (size_t)b * 65536;  // 16*4096
    for (int i = 0; i < 16; i++) {
        int idx = i * 256 + tid;
        int cl = idx >> 10, sp = idx & 1023;
        int oh = sp >> 5, ow = sp & 31;
        int co = g * 4 + cl;
        float acc = bias[co];
        for (int ci = 0; ci < 16; ci++) {
            const float* ip = ib + ci * 4096;
            const float* wp = w + (co * 16 + ci) * 16;
            #pragma unroll
            for (int kh = 0; kh < 4; kh++) {
                int ih = 2 * oh - 1 + kh;
                if (ih < 0 || ih >= 64) continue;
                #pragma unroll
                for (int kw = 0; kw < 4; kw++) {
                    int iw = 2 * ow - 1 + kw;
                    if (iw < 0 || iw >= 64) continue;
                    acc += ip[ih * 64 + iw] * wp[kh * 4 + kw];
                }
            }
        }
        vals[i] = acc; s += acc; q += acc * acc;
    }
    block_sum2(s, q);
    float mean = s * (1.f / 4096.f);
    float var  = q * (1.f / 4096.f) - mean * mean;
    float inv  = rsqrtf(var + 1e-5f);
    for (int i = 0; i < 16; i++) {
        int idx = i * 256 + tid;
        int cl = idx >> 10, sp = idx & 1023;
        int co = g * 4 + cl;
        float v = (vals[i] - mean) * inv * gam[co] + bet[co];
        out[((size_t)b * 32 + co) * 1024 + sp] = geluf(v);
    }
}

// ie3: [B,32,32,32] -> [B,64,16,16], GN groups=8 (8 ch/group)
__global__ void ie_conv3_kernel(const float* __restrict__ in, const float* __restrict__ w,
                                const float* __restrict__ bias, const float* __restrict__ gam,
                                const float* __restrict__ bet, float* __restrict__ out) {
    int b = blockIdx.x, g = blockIdx.y, tid = threadIdx.x;
    float vals[8]; float s = 0.f, q = 0.f;
    const float* ib = in + (size_t)b * 32768;  // 32*1024
    for (int i = 0; i < 8; i++) {
        int idx = i * 256 + tid;
        int cl = idx >> 8, sp = idx & 255;
        int oh = sp >> 4, ow = sp & 15;
        int co = g * 8 + cl;
        float acc = bias[co];
        for (int ci = 0; ci < 32; ci++) {
            const float* ip = ib + ci * 1024;
            const float* wp = w + (co * 32 + ci) * 16;
            #pragma unroll
            for (int kh = 0; kh < 4; kh++) {
                int ih = 2 * oh - 1 + kh;
                if (ih < 0 || ih >= 32) continue;
                #pragma unroll
                for (int kw = 0; kw < 4; kw++) {
                    int iw = 2 * ow - 1 + kw;
                    if (iw < 0 || iw >= 32) continue;
                    acc += ip[ih * 32 + iw] * wp[kh * 4 + kw];
                }
            }
        }
        vals[i] = acc; s += acc; q += acc * acc;
    }
    block_sum2(s, q);
    float mean = s * (1.f / 2048.f);
    float var  = q * (1.f / 2048.f) - mean * mean;
    float inv  = rsqrtf(var + 1e-5f);
    for (int i = 0; i < 8; i++) {
        int idx = i * 256 + tid;
        int cl = idx >> 8, sp = idx & 255;
        int co = g * 8 + cl;
        float v = (vals[i] - mean) * inv * gam[co] + bet[co];
        out[((size_t)b * 64 + co) * 256 + sp] = geluf(v);
    }
}

// ie4: [B,64,16,16] -> GN+GELU -> pool 8x8->4x4 -> z_image [B,16,128]
// GN groups=8 (16 ch/group)
__global__ void ie_conv4_pool_kernel(const float* __restrict__ in, const float* __restrict__ w,
                                     const float* __restrict__ bias, const float* __restrict__ gam,
                                     const float* __restrict__ bet, float* __restrict__ zimg) {
    int b = blockIdx.x, g = blockIdx.y, tid = threadIdx.x;
    __shared__ float pbuf[1024];  // 16 ch x 8x8 normalized values
    float vals[4]; float s = 0.f, q = 0.f;
    const float* ib = in + (size_t)b * 16384;  // 64*256
    for (int i = 0; i < 4; i++) {
        int idx = i * 256 + tid;
        int cl = idx >> 6, sp = idx & 63;
        int oh = sp >> 3, ow = sp & 7;
        int co = g * 16 + cl;
        float acc = bias[co];
        for (int ci = 0; ci < 64; ci++) {
            const float* ip = ib + ci * 256;
            const float* wp = w + (co * 64 + ci) * 16;
            #pragma unroll
            for (int kh = 0; kh < 4; kh++) {
                int ih = 2 * oh - 1 + kh;
                if (ih < 0 || ih >= 16) continue;
                #pragma unroll
                for (int kw = 0; kw < 4; kw++) {
                    int iw = 2 * ow - 1 + kw;
                    if (iw < 0 || iw >= 16) continue;
                    acc += ip[ih * 16 + iw] * wp[kh * 4 + kw];
                }
            }
        }
        vals[i] = acc; s += acc; q += acc * acc;
    }
    block_sum2(s, q);
    float mean = s * (1.f / 1024.f);
    float var  = q * (1.f / 1024.f) - mean * mean;
    float inv  = rsqrtf(var + 1e-5f);
    for (int i = 0; i < 4; i++) {
        int idx = i * 256 + tid;
        int cl = idx >> 6;
        int co = g * 16 + cl;
        float v = (vals[i] - mean) * inv * gam[co] + bet[co];
        pbuf[idx] = geluf(v);
    }
    __syncthreads();
    // pool 2x2 -> 4x4; one output per thread (16 ch * 16 positions)
    int cl = tid >> 4, pp = tid & 15;
    int ph = pp >> 2, pw = pp & 3;
    int base = cl * 64 + (ph * 2) * 8 + pw * 2;
    float avg = 0.25f * (pbuf[base] + pbuf[base + 1] + pbuf[base + 8] + pbuf[base + 9]);
    int co = g * 16 + cl;
    int tok = ph * 4 + pw;
    zimg[((size_t)b * 16 + tok) * 128 + co] = avg;
}

// ---------------- WMMA GEMM: out[m][n] = sum_k A[m][k]*W[n][k] + bias[n] ----------------
// block = 256 threads (8 waves in 4(M) x 2(N)); macro tile 64x32; K chunk 32.
// Tiles staged with CDNA5 async memory->LDS copies (ASYNCcnt).
__global__ void gemm_bias_kernel(const u16* __restrict__ A, const u16* __restrict__ W,
                                 const float* __restrict__ bias, float* __restrict__ out,
                                 int N, int K) {
    __shared__ __align__(16) u16 As[64 * 32];
    __shared__ __align__(16) u16 Ws[32 * 32];
    int gm0 = blockIdx.x * 64, n0 = blockIdx.y * 32;
    int tid = threadIdx.x, lane = tid & 31, wv = tid >> 5;
    int waveM = wv >> 1, waveN = wv & 1;
    int r = tid >> 2, cg = (tid & 3) << 3;
    v8f acc = {};
    for (int kc = 0; kc < K; kc += 32) {
        async_load_b128(As + r * 32 + cg, A + (size_t)(gm0 + r) * K + kc + cg);
        if (tid < 128)
            async_load_b128(Ws + r * 32 + cg, W + (size_t)(n0 + r) * K + kc + cg);
        if (kc + 32 < K)
            __builtin_prefetch(A + (size_t)(gm0 + r) * K + kc + 32 + cg, 0, 1);
        wait_async0();
        __syncthreads();
        acc = wmma_bf16_frag(As, 32, Ws, 32, 0, lane, waveM, waveN, acc);
        __syncthreads();
    }
    int half = lane >> 4, l16 = lane & 15;
    int n = n0 + waveN * 16 + l16;
    float bv = bias[n];
    #pragma unroll
    for (int rr = 0; rr < 8; rr++) {
        int m = gm0 + waveM * 16 + half * 8 + rr;
        out[(size_t)m * N + n] = acc[rr] + bv;
    }
}

// ---------------- attention (S=16 keys, VALU) ----------------
__global__ void attention_kernel(const float* __restrict__ qm, const float* __restrict__ kp,
                                 const float* __restrict__ vp, float* __restrict__ o) {
    int b = blockIdx.y, tid = threadIdx.x;
    __shared__ float Ks[2048], Vs[2048];
    for (int i = tid; i < 2048; i += 256) {
        Ks[i] = kp[(size_t)b * 2048 + i];
        Vs[i] = vp[(size_t)b * 2048 + i];
    }
    __syncthreads();
    int pair = blockIdx.x * 256 + tid;   // (m, h)
    int m = pair >> 2, h = pair & 3;
    const float* qr = qm + ((size_t)b * 1024 + m) * 128 + h * 32;
    float qv[32];
    #pragma unroll
    for (int d = 0; d < 32; d++) qv[d] = qr[d];
    float sc[16], mx = -3.0e38f;
    #pragma unroll
    for (int t = 0; t < 16; t++) {
        float a = 0.f;
        #pragma unroll
        for (int d = 0; d < 32; d++) a += qv[d] * Ks[t * 128 + h * 32 + d];
        a *= 0.17677669529663687f;  // 1/sqrt(32)
        sc[t] = a; mx = fmaxf(mx, a);
    }
    float den = 0.f;
    #pragma unroll
    for (int t = 0; t < 16; t++) { sc[t] = expf(sc[t] - mx); den += sc[t]; }
    float inv = 1.0f / den;
    float ov[32];
    #pragma unroll
    for (int d = 0; d < 32; d++) ov[d] = 0.f;
    #pragma unroll
    for (int t = 0; t < 16; t++) {
        float a = sc[t] * inv;
        #pragma unroll
        for (int d = 0; d < 32; d++) ov[d] += a * Vs[t * 128 + h * 32 + d];
    }
    float* orow = o + ((size_t)b * 1024 + m) * 128 + h * 32;
    #pragma unroll
    for (int d = 0; d < 32; d++) orow[d] = ov[d];
}

// ---------------- LayerNorm(z_motion + z_cross_pre), wave per row ----------------
__global__ void ln_kernel(const float* __restrict__ zm, const float* __restrict__ pre,
                          const float* __restrict__ g, const float* __restrict__ bta,
                          float* __restrict__ zc, u16* __restrict__ zcbf) {
    int row = blockIdx.x * 8 + (threadIdx.x >> 5);
    int lane = threadIdx.x & 31;
    const float* a = zm + (size_t)row * 128;
    const float* p = pre + (size_t)row * 128;
    float v[4]; float s = 0.f, q = 0.f;
    #pragma unroll
    for (int i = 0; i < 4; i++) {
        v[i] = a[lane * 4 + i] + p[lane * 4 + i];
        s += v[i]; q += v[i] * v[i];
    }
    #pragma unroll
    for (int off = 16; off > 0; off >>= 1) {
        s += __shfl_xor(s, off, 32);
        q += __shfl_xor(q, off, 32);
    }
    float mean = s * (1.f / 128.f);
    float var  = q * (1.f / 128.f) - mean * mean;
    float inv  = rsqrtf(var + 1e-5f);
    #pragma unroll
    for (int i = 0; i < 4; i++) {
        int c = lane * 4 + i;
        float y = (v[i] - mean) * inv * g[c] + bta[c];
        zc[(size_t)row * 128 + c]   = y;
        zcbf[(size_t)row * 128 + c] = f2bf(y);
    }
}

// ---------------- gated fusion: sigmoid(cat(zm,zc) @ Wg^T + b) blend ----------------
__global__ void gate_fuse_kernel(const u16* __restrict__ Azm, const u16* __restrict__ Azc,
                                 const u16* __restrict__ Wg, const float* __restrict__ gb,
                                 const float* __restrict__ zc, const float* __restrict__ zm,
                                 float* __restrict__ zout, u16* __restrict__ zbf) {
    __shared__ __align__(16) u16 As[64 * 32];
    __shared__ __align__(16) u16 Ws[32 * 32];
    int gm0 = blockIdx.x * 64, n0 = blockIdx.y * 32;
    int tid = threadIdx.x, lane = tid & 31, wv = tid >> 5;
    int waveM = wv >> 1, waveN = wv & 1;
    int r = tid >> 2, cg = (tid & 3) << 3;
    v8f acc = {};
    for (int kc = 0; kc < 256; kc += 32) {
        const u16* Asrc = (kc < 128) ? (Azm + kc) : (Azc + (kc - 128));
        async_load_b128(As + r * 32 + cg, Asrc + (size_t)(gm0 + r) * 128 + cg);
        if (tid < 128)
            async_load_b128(Ws + r * 32 + cg, Wg + (size_t)(n0 + r) * 256 + kc + cg);
        wait_async0();
        __syncthreads();
        acc = wmma_bf16_frag(As, 32, Ws, 32, 0, lane, waveM, waveN, acc);
        __syncthreads();
    }
    int half = lane >> 4, l16 = lane & 15;
    int n = n0 + waveN * 16 + l16;
    float bv = gb[n];
    #pragma unroll
    for (int rr = 0; rr < 8; rr++) {
        int m = gm0 + waveM * 16 + half * 8 + rr;
        float gg  = sigm(acc[rr] + bv);
        float zcv = zc[(size_t)m * 128 + n];
        float zmv = zm[(size_t)m * 128 + n];
        float z = gg * zcv + (1.0f - gg) * zmv;
        zout[(size_t)m * 128 + n] = z;
        zbf[(size_t)m * 128 + n]  = f2bf(z);
    }
}

// ---------------- codebook row norms ----------------
__global__ void enorm_kernel(const float* __restrict__ cb, float* __restrict__ en) {
    int row = blockIdx.x * 8 + (threadIdx.x >> 5);
    int lane = threadIdx.x & 31;
    float s = 0.f;
    #pragma unroll
    for (int i = 0; i < 4; i++) {
        float x = cb[(size_t)row * 128 + lane * 4 + i];
        s += x * x;
    }
    #pragma unroll
    for (int off = 16; off > 0; off >>= 1) s += __shfl_xor(s, off, 32);
    if (lane == 0) en[row] = s;
}

// ---------------- fused VQ: WMMA z@E^T, argmin over 2048 codes, gather, loss ----------------
// One block per 64 rows.  Never materializes the 65536x2048 score matrix.
__global__ void vq_kernel(const u16* __restrict__ Zbf, const u16* __restrict__ CBbf,
                          const float* __restrict__ Enorm, const float* __restrict__ Zf,
                          const float* __restrict__ CBf, float* __restrict__ qst,
                          int* __restrict__ idx_out, float* __restrict__ loss) {
    __shared__ __align__(16) u16 As[64 * 128];   // full 64-row A panel, K=128
    __shared__ __align__(16) u16 Ws[32 * 128];   // 32 codebook rows
    __shared__ float redV[64 * 2];
    __shared__ int   redI[64 * 2];
    __shared__ int   rowIdx[64];
    __shared__ float lm[8];
    int gm0 = blockIdx.x * 64;
    int tid = threadIdx.x, lane = tid & 31, wv = tid >> 5;
    int waveM = wv >> 1, waveN = wv & 1;
    int half = lane >> 4, l16 = lane & 15;

    #pragma unroll
    for (int i = 0; i < 4; i++) {   // stage A panel: 1024 async 16B copies
        int e = tid + i * 256;
        async_load_b128(As + e * 8, Zbf + (size_t)gm0 * 128 + e * 8);
    }
    float best[8]; int bidx[8];
    #pragma unroll
    for (int r = 0; r < 8; r++) { best[r] = 3.0e38f; bidx[r] = 0; }

    for (int nc = 0; nc < 64; nc++) {
        int n0 = nc * 32;
        #pragma unroll
        for (int i = 0; i < 2; i++) {  // stage 32 codebook rows (async)
            int e = tid + i * 256;
            async_load_b128(Ws + e * 8, CBbf + (size_t)n0 * 128 + e * 8);
        }
        wait_async0();
        __syncthreads();
        v8f acc = {};
        #pragma unroll
        for (int kc = 0; kc < 128; kc += 32)
            acc = wmma_bf16_frag(As, 128, Ws, 128, kc, lane, waveM, waveN, acc);
        int n = n0 + waveN * 16 + l16;
        float en = Enorm[n];
        #pragma unroll
        for (int r = 0; r < 8; r++) {
            float v = en - 2.0f * acc[r];
            if (v < best[r]) { best[r] = v; bidx[r] = n; }
        }
        __syncthreads();
    }
    // reduce across the 16 lanes of each N-group (rows stay within 16-lane halves)
    #pragma unroll
    for (int off = 1; off < 16; off <<= 1) {
        #pragma unroll
        for (int r = 0; r < 8; r++) {
            float ov = __shfl_xor(best[r], off, 32);
            int   oi = __shfl_xor(bidx[r], off, 32);
            if (ov < best[r] || (ov == best[r] && oi < bidx[r])) { best[r] = ov; bidx[r] = oi; }
        }
    }
    if (l16 == 0) {
        #pragma unroll
        for (int r = 0; r < 8; r++) {
            int row = waveM * 16 + half * 8 + r;
            redV[row * 2 + waveN] = best[r];
            redI[row * 2 + waveN] = bidx[r];
        }
    }
    __syncthreads();
    if (tid < 64) {
        float v0 = redV[tid * 2], v1 = redV[tid * 2 + 1];
        int   i0 = redI[tid * 2], i1 = redI[tid * 2 + 1];
        int ii = (v1 < v0 || (v1 == v0 && i1 < i0)) ? i1 : i0;
        rowIdx[tid] = ii;
        idx_out[gm0 + tid] = ii;
    }
    __syncthreads();
    // gather exact f32 codebook rows -> q_st, accumulate (z - quant)^2
    float lsum = 0.f;
    for (int i = 0; i < 32; i++) {
        int flat = tid * 32 + i;          // 64*128 elements
        int row = flat >> 7, col = flat & 127;
        int ci = rowIdx[row];
        float qv = CBf[(size_t)ci * 128 + col];
        float zv = Zf[((size_t)(gm0 + row)) * 128 + col];
        qst[((size_t)(gm0 + row)) * 128 + col] = qv;
        float d = zv - qv;
        lsum += d * d;
    }
    #pragma unroll
    for (int off = 16; off > 0; off >>= 1) lsum += __shfl_xor(lsum, off, 32);
    if (lane == 0) lm[wv] = lsum;
    __syncthreads();
    if (tid == 0) {
        float t = 0.f;
        #pragma unroll
        for (int i = 0; i < 8; i++) t += lm[i];
        atomicAdd(loss, t * (1.25f / 8388608.0f));  // cb_loss + 0.25*commit
    }
}

// ---------------- host launch ----------------

extern "C" void kernel_launch(void* const* d_in, const int* in_sizes, int n_in,
                              void* d_out, int out_size, void* d_ws, size_t ws_size,
                              hipStream_t stream) {
    (void)in_sizes; (void)n_in; (void)out_size; (void)ws_size;
    const float* imu   = (const float*)d_in[0];
    const float* timg  = (const float*)d_in[1];
    const float* me_w1 = (const float*)d_in[2],  *me_b1 = (const float*)d_in[3];
    const float* me_g1 = (const float*)d_in[4],  *me_bb1= (const float*)d_in[5];
    const float* me_w2 = (const float*)d_in[6],  *me_b2 = (const float*)d_in[7];
    const float* me_g2 = (const float*)d_in[8],  *me_bb2= (const float*)d_in[9];
    const float* me_w3 = (const float*)d_in[10], *me_b3 = (const float*)d_in[11];
    const float* me_g3 = (const float*)d_in[12], *me_bb3= (const float*)d_in[13];
    const float* ie_w1 = (const float*)d_in[14], *ie_b1 = (const float*)d_in[15];
    const float* ie_g1 = (const float*)d_in[16], *ie_bb1= (const float*)d_in[17];
    const float* ie_w2 = (const float*)d_in[18], *ie_b2 = (const float*)d_in[19];
    const float* ie_g2 = (const float*)d_in[20], *ie_bb2= (const float*)d_in[21];
    const float* ie_w3 = (const float*)d_in[22], *ie_b3 = (const float*)d_in[23];
    const float* ie_g3 = (const float*)d_in[24], *ie_bb3= (const float*)d_in[25];
    const float* ie_w4 = (const float*)d_in[26], *ie_b4 = (const float*)d_in[27];
    const float* ie_g4 = (const float*)d_in[28], *ie_bb4= (const float*)d_in[29];
    const float* wq = (const float*)d_in[30], *bq = (const float*)d_in[31];
    const float* wk = (const float*)d_in[32], *bk = (const float*)d_in[33];
    const float* wv = (const float*)d_in[34], *bv = (const float*)d_in[35];
    const float* wo = (const float*)d_in[36], *bo = (const float*)d_in[37];
    const float* ln_g = (const float*)d_in[38], *ln_b = (const float*)d_in[39];
    const float* gate_w = (const float*)d_in[40], *gate_b = (const float*)d_in[41];
    const float* codebook = (const float*)d_in[42];

    // workspace layout (bytes)
    char* ws = (char*)d_ws;
    constexpr size_t SLABF = 8388608;                       // 64*1024*128 floats
    constexpr size_t OFF_S0    = 0;
    constexpr size_t OFF_S1    = OFF_S0 + SLABF * 4;
    constexpr size_t OFF_S2    = OFF_S1 + SLABF * 4;
    constexpr size_t OFF_I1    = OFF_S2 + SLABF * 4;        // 64*16*64*64 f32
    constexpr size_t OFF_I2    = OFF_I1 + 4194304ull * 4;   // 64*32*32*32
    constexpr size_t OFF_I3    = OFF_I2 + 2097152ull * 4;   // 64*64*16*16
    constexpr size_t OFF_ZIMG  = OFF_I3 + 1048576ull * 4;   // 64*16*128
    constexpr size_t OFF_KP    = OFF_ZIMG + 131072ull * 4;
    constexpr size_t OFF_VP    = OFF_KP + 131072ull * 4;
    constexpr size_t OFF_EN    = OFF_VP + 131072ull * 4;    // 2048 f32
    constexpr size_t OFF_ZMBF  = OFF_EN + 2048ull * 4;      // bf16 mirrors
    constexpr size_t OFF_OBF   = OFF_ZMBF + SLABF * 2;
    constexpr size_t OFF_ZCBF  = OFF_OBF + SLABF * 2;
    constexpr size_t OFF_ZBF   = OFF_ZCBF + SLABF * 2;
    constexpr size_t OFF_ZIBF  = OFF_ZBF + SLABF * 2;       // 131072 bf16
    constexpr size_t OFF_WQBF  = OFF_ZIBF + 131072ull * 2;
    constexpr size_t OFF_WKBF  = OFF_WQBF + 16384ull * 2;
    constexpr size_t OFF_WVBF  = OFF_WKBF + 16384ull * 2;
    constexpr size_t OFF_WOBF  = OFF_WVBF + 16384ull * 2;
    constexpr size_t OFF_GWBF  = OFF_WOBF + 16384ull * 2;
    constexpr size_t OFF_CBBF  = OFF_GWBF + 32768ull * 2;   // 2048*128 bf16

    float* S0   = (float*)(ws + OFF_S0);
    float* S1   = (float*)(ws + OFF_S1);
    float* S2   = (float*)(ws + OFF_S2);     // z_motion
    float* I1   = (float*)(ws + OFF_I1);
    float* I2   = (float*)(ws + OFF_I2);
    float* I3   = (float*)(ws + OFF_I3);
    float* ZIMG = (float*)(ws + OFF_ZIMG);
    float* KP   = (float*)(ws + OFF_KP);
    float* VP   = (float*)(ws + OFF_VP);
    float* EN   = (float*)(ws + OFF_EN);
    u16* ZMBF = (u16*)(ws + OFF_ZMBF);
    u16* OBF  = (u16*)(ws + OFF_OBF);
    u16* ZCBF = (u16*)(ws + OFF_ZCBF);
    u16* ZBF  = (u16*)(ws + OFF_ZBF);
    u16* ZIBF = (u16*)(ws + OFF_ZIBF);
    u16* WQBF = (u16*)(ws + OFF_WQBF);
    u16* WKBF = (u16*)(ws + OFF_WKBF);
    u16* WVBF = (u16*)(ws + OFF_WVBF);
    u16* WOBF = (u16*)(ws + OFF_WOBF);
    u16* GWBF = (u16*)(ws + OFF_GWBF);
    u16* CBBF = (u16*)(ws + OFF_CBBF);

    float* out_qst  = (float*)d_out;                 // 8388608
    float* out_loss = (float*)d_out + 8388608;       // 1
    int*   out_idx  = (int*)d_out + 8388609;         // 65536

    init_loss_kernel<<<1, 1, 0, stream>>>(out_loss);

    // weight / codebook bf16 conversion + codebook norms
    f32_to_bf16_kernel<<<64, 256, 0, stream>>>(wq, WQBF, 16384);
    f32_to_bf16_kernel<<<64, 256, 0, stream>>>(wk, WKBF, 16384);
    f32_to_bf16_kernel<<<64, 256, 0, stream>>>(wv, WVBF, 16384);
    f32_to_bf16_kernel<<<64, 256, 0, stream>>>(wo, WOBF, 16384);
    f32_to_bf16_kernel<<<128, 256, 0, stream>>>(gate_w, GWBF, 32768);
    f32_to_bf16_kernel<<<1024, 256, 0, stream>>>(codebook, CBBF, 262144);
    enorm_kernel<<<256, 256, 0, stream>>>(codebook, EN);

    // motion encoder
    me_conv1_kernel<<<dim3(64, 8), 256, 0, stream>>>(imu, me_w1, me_b1, me_g1, me_bb1, S0);
    me_conv2_kernel<<<dim3(64, 8), 256, 0, stream>>>(S0, me_w2, me_b2, me_g2, me_bb2, S1);
    me_conv3_kernel<<<dim3(64, 8), 256, 0, stream>>>(S1, me_w3, me_b3, me_g3, me_bb3, S2);

    // image encoder
    ie_conv1_kernel<<<dim3(64, 4), 256, 0, stream>>>(timg, ie_w1, ie_b1, ie_g1, ie_bb1, I1);
    ie_conv2_kernel<<<dim3(64, 8), 256, 0, stream>>>(I1, ie_w2, ie_b2, ie_g2, ie_bb2, I2);
    ie_conv3_kernel<<<dim3(64, 8), 256, 0, stream>>>(I2, ie_w3, ie_b3, ie_g3, ie_bb3, I3);
    ie_conv4_pool_kernel<<<dim3(64, 8), 256, 0, stream>>>(I3, ie_w4, ie_b4, ie_g4, ie_bb4, ZIMG);

    // activation bf16 mirrors
    f32_to_bf16_kernel<<<32768, 256, 0, stream>>>(S2, ZMBF, 8388608);
    f32_to_bf16_kernel<<<512, 256, 0, stream>>>(ZIMG, ZIBF, 131072);

    // projections (WMMA)
    gemm_bias_kernel<<<dim3(1024, 4), 256, 0, stream>>>(ZMBF, WQBF, bq, S0, 128, 128);  // Q
    gemm_bias_kernel<<<dim3(16, 4), 256, 0, stream>>>(ZIBF, WKBF, bk, KP, 128, 128);    // K
    gemm_bias_kernel<<<dim3(16, 4), 256, 0, stream>>>(ZIBF, WVBF, bv, VP, 128, 128);    // V

    // attention -> O (S1), then Wo projection -> z_cross_pre (S0)
    attention_kernel<<<dim3(16, 64), 256, 0, stream>>>(S0, KP, VP, S1);
    f32_to_bf16_kernel<<<32768, 256, 0, stream>>>(S1, OBF, 8388608);
    gemm_bias_kernel<<<dim3(1024, 4), 256, 0, stream>>>(OBF, WOBF, bo, S0, 128, 128);

    // residual + LN -> z_cross (S1) (+ bf16)
    ln_kernel<<<8192, 256, 0, stream>>>(S2, S0, ln_g, ln_b, S1, ZCBF);

    // gated fusion -> z (S0) (+ bf16)
    gate_fuse_kernel<<<dim3(1024, 4), 256, 0, stream>>>(ZMBF, ZCBF, GWBF, gate_b,
                                                        S1, S2, S0, ZBF);

    // fused VQ: argmin + gather + loss
    vq_kernel<<<1024, 256, 0, stream>>>(ZBF, CBBF, EN, S0, codebook,
                                        out_qst, out_idx, out_loss);
}
